// Xhead_2774548873523
// MI455X (gfx1250) — matrix operand order repaired and checked by reading, not compiled, not run on hardware
//
#include <hip/hip_runtime.h>

// ---------------------------------------------------------------------------
// MI455X (gfx1250) YOLOX-head pipeline.
// All 1x1 convs lower to v_wmma_f32_16x16x32_bf16 (bf16 operands, fp32 acc).
// Weights are pre-converted to bf16 once; A fragments load as raw b128;
// B fragments gather from LDS staged in [p][k] order as 2x ds_load_b128.
// Epilogues fuse bias+SiLU / bias+residual / full box decode + transpose.
// ---------------------------------------------------------------------------

typedef __attribute__((ext_vector_type(16))) __bf16         v16bf;
typedef __attribute__((ext_vector_type(16))) unsigned short v16us;
typedef __attribute__((ext_vector_type(8)))  float          v8f;

union BFV { v16us u; v16bf v; uint4 q[2]; };
union U8  { uint4 q; unsigned short s[8]; };

__device__ __forceinline__ unsigned short f2bf(float f) {
  unsigned int u = __float_as_uint(f);
  u += 0x7FFFu + ((u >> 16) & 1u);          // round-to-nearest-even
  return (unsigned short)(u >> 16);
}
__device__ __forceinline__ float bf2f(unsigned short h) {
  return __uint_as_float(((unsigned int)h) << 16);
}

// MODE: 1 = bias+SiLU -> bf16   2 = bias+residual -> bf16
//       3 = cls head: sigmoid, scatter to out[b,p,5+co]
//       4 = reg/obj head: decode, scatter to out[b,p,0..4]
// XF32: X operand is fp32 (stem reads raw inputs) else bf16
// MT:   number of 16-row output-channel tiles computed per wave
template <int MODE, bool XF32, int MT>
__global__ void __launch_bounds__(32)
gemm_wmma_head(const unsigned short* __restrict__ A,  // [MT*16, K] bf16 weights
               const void*  __restrict__ Xv,          // [B, K, HW] fp32 or bf16
               const float* __restrict__ bias,
               const unsigned short* __restrict__ res, // bf16 [B,96,HW] (MODE 2)
               unsigned short* __restrict__ outB,
               float* __restrict__ outF,
               int K, int HW, int Wimg, float sS, int posOff) {
  __shared__ unsigned short tl[16 * 32];     // [p][k] : 16 positions x 32 K bf16
  const int lane = (int)threadIdx.x;
  const int p0   = (int)blockIdx.x * 16;
  const int b    = (int)blockIdx.z;
  const long long xbs = (long long)K * HW;

  v8f acc[MT];
#pragma unroll
  for (int m = 0; m < MT; ++m)
#pragma unroll
    for (int i = 0; i < 8; ++i) acc[m][i] = 0.f;

  const int rowA = lane & 15;
  const int kbA  = (lane >= 16) ? 8 : 0;   // A: interleaved 8-blocks (ISA 7.12.2)
  const int colB = lane & 15;
  const int kbB  = (lane >= 16) ? 8 : 0;   // B: interleaved 8-blocks (matches A)

  for (int k0 = 0; k0 < K; k0 += 32) {
    // ---- stage one 32K x 16P activation tile into LDS, transposed [p][k] ----
    if (XF32) {
      const float* xr = (const float*)Xv + (long long)b * xbs +
                        (long long)(k0 + lane) * HW + p0;
      const float4* px = (const float4*)xr;
#pragma unroll
      for (int j = 0; j < 4; ++j) {
        float4 f = px[j];
        tl[(4 * j + 0) * 32 + lane] = f2bf(f.x);
        tl[(4 * j + 1) * 32 + lane] = f2bf(f.y);
        tl[(4 * j + 2) * 32 + lane] = f2bf(f.z);
        tl[(4 * j + 3) * 32 + lane] = f2bf(f.w);
      }
    } else {
      const unsigned short* xr = (const unsigned short*)Xv + (long long)b * xbs +
                                 (long long)(k0 + lane) * HW + p0;
      const uint4* px = (const uint4*)xr;
      U8 q0, q1; q0.q = px[0]; q1.q = px[1];
#pragma unroll
      for (int j = 0; j < 8; ++j) {
        tl[j * 32 + lane]       = q0.s[j];
        tl[(8 + j) * 32 + lane] = q1.s[j];
      }
    }
    __syncthreads();

    // ---- B fragment: 2x ds_load_b128 per lane (k blocks kbB, kbB+16) ----
    BFV vb;
    {
      const uint4* bp = (const uint4*)&tl[colB * 32 + kbB];
      vb.q[0] = bp[0];
      vb.q[1] = bp[2];
    }

    // ---- A fragments: raw bf16 b128 loads, then WMMA ----
#pragma unroll
    for (int m = 0; m < MT; ++m) {
      const unsigned short* ar = A + (long long)(m * 16 + rowA) * K + k0 + kbA;
      BFV va;
      va.q[0] = *(const uint4*)(ar);
      va.q[1] = *(const uint4*)(ar + 16);
      acc[m] = __builtin_amdgcn_wmma_f32_16x16x32_bf16(
          false, va.v, false, vb.v, (short)0, acc[m], false, false);
    }
    __syncthreads();
  }

  // ---- fused epilogue; C/D layout: VGPR v -> M = v + 8*(lane>=16), N = lane&15
  const int hi = lane >> 4;
  const int pl = lane & 15;
  const int p  = p0 + pl;
  const long long obs = (long long)MT * 16 * HW;
#pragma unroll
  for (int m = 0; m < MT; ++m) {
#pragma unroll
    for (int v = 0; v < 8; ++v) {
      const int co = m * 16 + v + 8 * hi;
      float x = acc[m][v] + bias[co];
      if (MODE == 1) {
        float s = x / (1.f + __expf(-x));                       // SiLU
        outB[(long long)b * obs + (long long)co * HW + p] = f2bf(s);
      } else if (MODE == 2) {
        x += bf2f(res[(long long)b * 96 * HW + (long long)co * HW + p]);
        outB[(long long)b * obs + (long long)co * HW + p] = f2bf(x);
      } else if (MODE == 3) {
        outF[(long long)b * 714000 + (long long)(posOff + p) * 85 + 5 + co] =
            1.f / (1.f + __expf(-x));
      } else {                                                  // MODE 4
        if (co < 5) {
          float val;
          if (co == 0)      val = (x + (float)(p % Wimg)) * sS;
          else if (co == 1) val = (x + (float)(p / Wimg)) * sS;
          else if (co == 4) val = 1.f / (1.f + __expf(-x));
          else              val = __expf(x) * sS;
          outF[(long long)b * 714000 + (long long)(posOff + p) * 85 + co] = val;
        }
      }
    }
  }
}

// 3x3 depthwise conv (pad 1) + SiLU, bf16 in/out, fp32 math.
__global__ void __launch_bounds__(256)
dw3x3_silu(const unsigned short* __restrict__ in, const float* __restrict__ w,
           const float* __restrict__ bias, unsigned short* __restrict__ out,
           int C, int H, int W, long long total) {
  long long idx = (long long)blockIdx.x * 256 + threadIdx.x;
  if (idx >= total) return;
  int xw = (int)(idx % W);
  long long t = idx / W;
  int yh = (int)(t % H); t /= H;
  int c  = (int)(t % C);
  int b  = (int)(t / C);
  const float* wc = w + c * 9;
  const unsigned short* ip = in + ((long long)b * C + c) * H * W;
  float acc = bias[c];
#pragma unroll
  for (int dy = 0; dy < 3; ++dy) {
    int yy = yh + dy - 1;
    if (yy < 0 || yy >= H) continue;
#pragma unroll
    for (int dx = 0; dx < 3; ++dx) {
      int xx = xw + dx - 1;
      if (xx < 0 || xx >= W) continue;
      acc += bf2f(ip[(long long)yy * W + xx]) * wc[dy * 3 + dx];
    }
  }
  float s = acc / (1.f + __expf(-acc));
  out[idx] = f2bf(s);
}

// Generic fp32 -> bf16 weight conversion (run once per launch; weights ~0.7MB).
__global__ void __launch_bounds__(256)
f32_to_bf16_k(const float* __restrict__ src, unsigned short* __restrict__ dst, int n) {
  int i = (int)blockIdx.x * 256 + (int)threadIdx.x;
  if (i < n) dst[i] = f2bf(src[i]);
}

// Pack reg(4)+obj(1) head into one zero-padded [16,96] bf16 weight + [16] f32 bias.
__global__ void __launch_bounds__(256)
pack_regobj(const float* __restrict__ regw, const float* __restrict__ objw,
            const float* __restrict__ regb, const float* __restrict__ objb,
            unsigned short* __restrict__ rw, float* __restrict__ rb) {
  int k = blockIdx.x;
  for (int i = threadIdx.x; i < 16 * 96; i += 256) {
    int row = i / 96, col = i % 96;
    float v = 0.f;
    if (row < 4)       v = regw[(k * 4 + row) * 96 + col];
    else if (row == 4) v = objw[k * 96 + col];
    rw[k * 16 * 96 + i] = f2bf(v);
  }
  if (threadIdx.x < 16) {
    int row = threadIdx.x;
    float v = 0.f;
    if (row < 4)       v = regb[k * 4 + row];
    else if (row == 4) v = objb[k];
    rb[k * 16 + row] = v;
  }
}

extern "C" void kernel_launch(void* const* d_in, const int* in_sizes, int n_in,
                              void* d_out, int out_size, void* d_ws, size_t ws_size,
                              hipStream_t stream) {
  (void)in_sizes; (void)n_in; (void)out_size; (void)ws_size;

  static const int   HWa[3]  = {6400, 1600, 400};
  static const int   Wa[3]   = {80, 40, 20};
  static const int   Cin[3]  = {128, 256, 512};
  static const float Sa[3]   = {8.f, 16.f, 32.f};
  static const int   Poff[3] = {0, 6400, 8000};

  const float* xk[3]  = {(const float*)d_in[0], (const float*)d_in[3], (const float*)d_in[6]};
  const float* swk[3] = {(const float*)d_in[1], (const float*)d_in[4], (const float*)d_in[7]};
  const float* sbk[3] = {(const float*)d_in[2], (const float*)d_in[5], (const float*)d_in[8]};
  const float* w1[2]  = {(const float*)d_in[9],  (const float*)d_in[15]};
  const float* b1[2]  = {(const float*)d_in[10], (const float*)d_in[16]};
  const float* wdp[2] = {(const float*)d_in[11], (const float*)d_in[17]};
  const float* bdp[2] = {(const float*)d_in[12], (const float*)d_in[18]};
  const float* w2[2]  = {(const float*)d_in[13], (const float*)d_in[19]};
  const float* b2[2]  = {(const float*)d_in[14], (const float*)d_in[20]};
  const float* cpw = (const float*)d_in[21];
  const float* cpb = (const float*)d_in[22];

  // ---- workspace layout (regions written before read on every call) ----
  char* ws = (char*)d_ws;
  // bf16 weight pool (front 1 MB)
  unsigned short* stw[3] = {(unsigned short*)(ws + 0),
                            (unsigned short*)(ws + 24576),
                            (unsigned short*)(ws + 73728)};
  unsigned short* w1b[2] = {(unsigned short*)(ws + 172032),
                            (unsigned short*)(ws + 282624)};
  unsigned short* w2b[2] = {(unsigned short*)(ws + 393216),
                            (unsigned short*)(ws + 503808)};
  unsigned short* cpwb   = (unsigned short*)(ws + 614400);
  unsigned short* rob    = (unsigned short*)(ws + 660480);
  float*          rbf    = (float*)(ws + 669696);
  // bf16 activation buffers
  unsigned short* stem = (unsigned short*)(ws + 1048576);
  unsigned short* feat = (unsigned short*)(ws + 1048576 + 19660800);
  unsigned short* expb = (unsigned short*)(ws + 1048576 + 2 * 19660800);
  unsigned short* dwb  = (unsigned short*)(ws + 1048576 + 2 * 19660800 + 39321600);
  float* out = (float*)d_out;

  // ---- one-time (per launch) weight conversion to bf16 ----
  dim3 cb(256);
  for (int k = 0; k < 3; ++k) {
    int n = 96 * Cin[k];
    f32_to_bf16_k<<<dim3((n + 255) / 256), cb, 0, stream>>>(swk[k], stw[k], n);
  }
  for (int br = 0; br < 2; ++br) {
    f32_to_bf16_k<<<dim3((3 * 192 * 96 + 255) / 256), cb, 0, stream>>>(w1[br], w1b[br], 3 * 192 * 96);
    f32_to_bf16_k<<<dim3((3 * 96 * 192 + 255) / 256), cb, 0, stream>>>(w2[br], w2b[br], 3 * 96 * 192);
  }
  f32_to_bf16_k<<<dim3((3 * 80 * 96 + 255) / 256), cb, 0, stream>>>(cpw, cpwb, 3 * 80 * 96);
  pack_regobj<<<dim3(3), cb, 0, stream>>>(
      (const float*)d_in[23], (const float*)d_in[25],
      (const float*)d_in[24], (const float*)d_in[26], rob, rbf);

  // ---- per-level pipeline ----
  for (int k = 0; k < 3; ++k) {
    const int HW = HWa[k], W = Wa[k];
    dim3 g(HW / 16, 1, 16), blk(32);

    // stem 1x1: Cin->96, SiLU (reads fp32 inputs directly, converts in staging)
    gemm_wmma_head<1, true, 6><<<g, blk, 0, stream>>>(
        stw[k], xk[k], sbk[k], nullptr, stem, nullptr, Cin[k], HW, W, 0.f, 0);

    for (int br = 0; br < 2; ++br) {   // 0 = cls branch, 1 = reg branch
      // expand 1x1: 96->192, SiLU
      gemm_wmma_head<1, false, 12><<<g, blk, 0, stream>>>(
          w1b[br] + k * 192 * 96, stem, b1[br] + k * 192, nullptr, expb, nullptr,
          96, HW, W, 0.f, 0);
      // 3x3 depthwise + SiLU
      long long tot = 16LL * 192 * HW;
      dw3x3_silu<<<dim3((unsigned)((tot + 255) / 256)), dim3(256), 0, stream>>>(
          expb, wdp[br] + k * 192 * 9, bdp[br] + k * 192, dwb, 192, W, W, tot);
      // project 1x1: 192->96, + residual(stem)
      gemm_wmma_head<2, false, 6><<<g, blk, 0, stream>>>(
          w2b[br] + k * 96 * 192, dwb, b2[br] + k * 96, stem, feat, nullptr,
          192, HW, W, 0.f, 0);
      if (br == 0) {
        // cls head 96->80, sigmoid, scatter to out[b, p, 5..84]
        gemm_wmma_head<3, false, 5><<<g, blk, 0, stream>>>(
            cpwb + k * 80 * 96, feat, cpb + k * 80, nullptr, nullptr, out,
            96, HW, W, 0.f, Poff[k]);
      } else {
        // packed reg+obj head 96->16(5 live), decode, scatter to out[b, p, 0..4]
        gemm_wmma_head<4, false, 1><<<g, blk, 0, stream>>>(
            rob + k * 16 * 96, feat, rbf + k * 16, nullptr, nullptr, out,
            96, HW, W, Sa[k], Poff[k]);
      }
    }
  }
}
